// SepConvNet_5437428597341
// MI455X (gfx1250) — matrix operand order repaired and checked by reading, not compiled, not run on hardware
//
#include <hip/hip_runtime.h>
#include <hip/hip_bf16.h>

// SepConv (adaptive per-pixel separable conv) for MI455X / gfx1250.
//
// Strategy: per block = 16 output cols x 8 output rows (one row per wave32),
// fixed batch b. Horizontal pass as a banded GEMM on v_wmma_f32_16x16x32_f16:
//   tmp[i, x^] = sum_j' A[i, j'] * Hsh[j', x^]
//   A[i,j']    = clamped_frame[y-25+i, x0-25+j']      (f16, staged in LDS)
//   Hsh[j',x^] = (0 <= j'-x^ < 51) ? H[j'-x^, y, x0+x^] : 0   (f16, registers)
// i padded 51->64 (4 M-tiles), j' padded 66->96 (3 K-steps of 32).
// Vertical pass + frame-pair sum in fp32 VALU directly on the WMMA C layout.
// V/H arrays (428 MB) stream from HBM exactly once; frames (12.6 MB) are
// L2-resident so the halo re-reads are cheap.

typedef __attribute__((ext_vector_type(16))) _Float16 v16h;
typedef __attribute__((ext_vector_type(8)))  _Float16 v8h;
typedef __attribute__((ext_vector_type(8)))  float    v8f;

#define Bc   2
#define Cc   3
#define Hc   512
#define Wc   512
#define Kc   51
#define PADc 25
#define NX   16            // output columns per block (one WMMA N-tile)
#define NY   8             // output rows per block (one per wave)
#define A_ROWS   72        // NY-1 + 64 M-rows = 71 -> 72
#define A_COLS   96        // NX-1 + K = 66 -> 3 K-steps of 32
#define A_STRIDE 104       // halves: 208B rows, 16B aligned, bank-spread

__global__ __launch_bounds__(256) void sepconv_wmma(
    const float* __restrict__ f0, const float* __restrict__ f2,
    const float* __restrict__ V1, const float* __restrict__ H1,
    const float* __restrict__ V2, const float* __restrict__ H2,
    float* __restrict__ out)
{
  __shared__ __align__(16) _Float16 As[Cc][A_ROWS][A_STRIDE];

  const int x0 = blockIdx.x * NX;
  const int y0 = blockIdx.y * NY;
  const int b  = blockIdx.z;

  const int tid  = threadIdx.x;
  const int wave = tid >> 5;      // gfx1250 is wave32
  const int lane = tid & 31;
  const int ln   = lane & 15;     // N index (output column within tile)
  const int lh   = lane >> 4;     // lane-half select
  const int y    = y0 + wave;     // this wave's output row

  float sc[Cc] = {0.f, 0.f, 0.f};

  for (int f = 0; f < 2; ++f) {
    const float* Fp = f ? f2 : f0;
    const float* Hp = f ? H2 : H1;
    const float* Vp = f ? V2 : V1;

    __syncthreads();   // previous frame's tile fully consumed
    // ---- Stage clamped (replication-padded) frame tile, f32 -> f16 ----
    for (int idx = tid; idx < Cc * A_ROWS * A_COLS; idx += 256) {
      int cc = idx % A_COLS;
      int t  = idx / A_COLS;
      int rr = t % A_ROWS;
      int c  = t / A_ROWS;
      int gy = y0 - PADc + rr; gy = gy < 0 ? 0 : (gy > Hc - 1 ? Hc - 1 : gy);
      int gx = x0 - PADc + cc; gx = gx < 0 ? 0 : (gx > Wc - 1 ? Wc - 1 : gx);
      As[c][rr][cc] = (_Float16)Fp[((size_t)(b * Cc + c) * Hc + gy) * Wc + gx];
    }
    __syncthreads();

    // ---- Build banded Hsh B-fragments (32x16 f16, 3 K-steps) ----
    // B layout: lane l -> N = l%16; half h<8 -> K = (l/16)*8 + h, h>=8 -> K = 16 + (l/16)*8 + (h-8)
    v16h bfrag[3];
    #pragma unroll
    for (int ks = 0; ks < 3; ++ks) {
      #pragma unroll
      for (int h = 0; h < 16; ++h) {
        int jp = ks * 32 + lh * 8 + h + (h >= 8 ? 8 : 0);   // absolute j'
        int j  = jp - ln;                 // horizontal tap index
        float v = 0.f;
        if (j >= 0 && j < Kc)
          v = Hp[((size_t)(b * Kc + j) * Hc + y) * Wc + x0 + ln];
        bfrag[ks][h] = (_Float16)v;
      }
    }

    // ---- Load V in the WMMA C layout (zeroed for padded i >= 51) ----
    // C layout: lane l, reg r -> M = r + 8*(l/16), N = l%16
    float vw[4][8];
    #pragma unroll
    for (int mt = 0; mt < 4; ++mt) {
      #pragma unroll
      for (int r = 0; r < 8; ++r) {
        int i = mt * 16 + lh * 8 + r;
        vw[mt][r] = (i < Kc)
            ? Vp[((size_t)(b * Kc + i) * Hc + y) * Wc + x0 + ln] : 0.f;
      }
    }

    // ---- Per channel: banded GEMM (12 WMMAs) + vertical contraction ----
    for (int c = 0; c < Cc; ++c) {
      v8f acc[4];
      #pragma unroll
      for (int mt = 0; mt < 4; ++mt)
        #pragma unroll
        for (int e = 0; e < 8; ++e)
          acc[mt][e] = 0.f;

      #pragma unroll
      for (int ks = 0; ks < 3; ++ks) {
        #pragma unroll
        for (int mt = 0; mt < 4; ++mt) {
          // A layout: lane l -> M = l%16; halves 0..7 -> K = (l/16)*8 + 0..7,
          // halves 8..15 -> K = 16 + (l/16)*8 + 0..7  => two b128 LDS reads.
          const int arow = wave + mt * 16 + ln;
          const _Float16* rowp = &As[c][arow][ks * 32 + lh * 8];
          union { v8h h8[2]; v16h h16; } ua;
          ua.h8[0] = *reinterpret_cast<const v8h*>(rowp);
          ua.h8[1] = *reinterpret_cast<const v8h*>(rowp + 16);
          acc[mt] = __builtin_amdgcn_wmma_f32_16x16x32_f16(
              false, ua.h16, false, bfrag[ks], (short)0, acc[mt],
              false, false);
        }
      }

      // out[x^] = sum_i V[i,x^] * tmp[i,x^]  (fp32, in C layout)
      float s = 0.f;
      #pragma unroll
      for (int mt = 0; mt < 4; ++mt)
        #pragma unroll
        for (int r = 0; r < 8; ++r)
          s += acc[mt][r] * vw[mt][r];
      sc[c] += s;   // accumulates across both frame pairs
    }
  }

  // Lanes l and l+16 hold the same N with disjoint M sets: one xor-16 add.
  #pragma unroll
  for (int c = 0; c < Cc; ++c) {
    float t = sc[c] + __shfl_xor(sc[c], 16, 32);
    if (lane < 16)
      out[((size_t)(b * Cc + c) * Hc + y) * Wc + x0 + lane] = t;
  }
}

extern "C" void kernel_launch(void* const* d_in, const int* in_sizes, int n_in,
                              void* d_out, int out_size, void* d_ws, size_t ws_size,
                              hipStream_t stream) {
  const float* f0 = (const float*)d_in[0];
  const float* f2 = (const float*)d_in[1];
  const float* V1 = (const float*)d_in[2];
  const float* H1 = (const float*)d_in[3];
  const float* V2 = (const float*)d_in[4];
  const float* H2 = (const float*)d_in[5];
  float* outp = (float*)d_out;
  dim3 grid(Wc / NX, Hc / NY, Bc);   // 32 x 64 x 2 blocks, 256 threads each
  sepconv_wmma<<<grid, 256, 0, stream>>>(f0, f2, V1, H1, V2, H2, outp);
  (void)in_sizes; (void)n_in; (void)out_size; (void)d_ws; (void)ws_size;
}
